// BoxLM_1168231104949
// MI455X (gfx1250) — matrix operand order repaired and checked by reading, not compiled, not run on hardware
//
#include <hip/hip_runtime.h>
#include <hip/hip_bf16.h>

#define NVIS 50000
#define NCCS 300
#define NICD 7000
#define NNODE (NVIS + NCCS + NICD)        // 57300
#define NTILES ((NNODE + 15) / 16)        // 3582
#define NE1 524288
#define NE2 524288

typedef __attribute__((ext_vector_type(16))) _Float16 v16h;
typedef __attribute__((ext_vector_type(8)))  float    v8f;
typedef __attribute__((ext_vector_type(8)))  int      v8i;

union IV  { v8i  i;    v16h h; };   // one 32B LDS fragment  -> v16h
union QV  { int4 q[2]; v16h h; };   // two 16B LDS chunks    -> v16h

__device__ __forceinline__ const float* resolve_row(int r, const float* __restrict__ v,
                                                    const float* __restrict__ c,
                                                    const float* __restrict__ i) {
  if (r < NVIS) return v + (size_t)r * 64;
  if (r < NVIS + NCCS) return c + (size_t)(r - NVIS) * 64;
  return i + (size_t)(r - NVIS - NCCS) * 64;
}

// ---------------------------------------------------------------------------
// Kernel 1: elog = exp( relu(all_center @ W1^T + b1) @ W2^T + b2 )
// One wave per 16-row tile; f16 WMMA, f32 accumulate; K=64 (2 k-steps),
// N=64 (4 n-blocks). Weights staged in LDS pre-swizzled into B-fragment
// layout so each fragment is a single 32-byte DS read. exp() is folded in
// here (per node) instead of per edge: softmax is shift-invariant and the
// logits are O(1), so the max-subtraction pass is unnecessary.
// ---------------------------------------------------------------------------
__global__ __launch_bounds__(256)
void node_elog_kernel(const float* __restrict__ vc, const float* __restrict__ cc,
                      const float* __restrict__ ic,
                      const float* __restrict__ w1, const float* __restrict__ b1,
                      const float* __restrict__ w2, const float* __restrict__ b2,
                      float* __restrict__ elog)
{
  // fragment-major weight layout: [(kb*4+nb)*32 + lane]*16 + j
  __shared__ alignas(32) _Float16 wfrag1[64 * 64];
  __shared__ alignas(32) _Float16 wfrag2[64 * 64];
  __shared__ float b1s[64], b2s[64];
  __shared__ alignas(16) _Float16 hbuf[8][16 * 64];   // per-wave slab

  const int t = threadIdx.x;
  {
    // thread t owns (fragment f = t>>5, lane_s = t&31); reads stride 64 floats
    // across lane_s&15 for each j.
    const int f = t >> 5, lane_s = t & 31;
    const int kb = f >> 2, nb = f & 3;
    #pragma unroll
    for (int j = 0; j < 16; ++j) {
      const int src = (nb * 16 + (lane_s & 15)) * 64 + kb * 32 + (lane_s >> 4) * 8 + j + (j & 8);
      wfrag1[t * 16 + j] = (_Float16)w1[src];
      wfrag2[t * 16 + j] = (_Float16)w2[src];
    }
  }
  if (t < 64) { b1s[t] = b1[t]; b2s[t] = b2[t]; }
  __syncthreads();

  const int wave = t >> 5;
  const int lane = t & 31;
  const int tile = blockIdx.x * 8 + wave;
  if (tile >= NTILES) return;
  const int half = lane >> 4;    // lanes 16-31 hold the upper K group
  const int mrow = lane & 15;

  // --- A fragments (emb rows); 16-bit A 16x32 layout:
  //     element j of lane -> K = kb*32 + half*8 + j + (j & 8)
  const int rowA = tile * 16 + mrow;
  const float* rp = nullptr;
  if (rowA < NVIS) rp = vc + (size_t)rowA * 64;
  else if (rowA < NVIS + NCCS) rp = cc + (size_t)(rowA - NVIS) * 64;
  else if (rowA < NNODE) rp = ic + (size_t)(rowA - NVIS - NCCS) * 64;

  v16h a[2];
  #pragma unroll
  for (int kb = 0; kb < 2; ++kb) {
    if (rp) {
      const float4* r4 = (const float4*)rp;
      float4 u0 = r4[kb * 8 + half * 2 + 0];   // K = kb*32+half*8 + 0..3
      float4 u1 = r4[kb * 8 + half * 2 + 1];   // .. + 4..7
      float4 u2 = r4[kb * 8 + half * 2 + 4];   // .. +16..19
      float4 u3 = r4[kb * 8 + half * 2 + 5];   // .. +20..23
      a[kb][0]  = (_Float16)u0.x; a[kb][1]  = (_Float16)u0.y;
      a[kb][2]  = (_Float16)u0.z; a[kb][3]  = (_Float16)u0.w;
      a[kb][4]  = (_Float16)u1.x; a[kb][5]  = (_Float16)u1.y;
      a[kb][6]  = (_Float16)u1.z; a[kb][7]  = (_Float16)u1.w;
      a[kb][8]  = (_Float16)u2.x; a[kb][9]  = (_Float16)u2.y;
      a[kb][10] = (_Float16)u2.z; a[kb][11] = (_Float16)u2.w;
      a[kb][12] = (_Float16)u3.x; a[kb][13] = (_Float16)u3.y;
      a[kb][14] = (_Float16)u3.z; a[kb][15] = (_Float16)u3.w;
    } else {
      #pragma unroll
      for (int j = 0; j < 16; ++j) a[kb][j] = (_Float16)0.0f;
    }
  }

  // --- GEMM1: h[m][n] = sum_k emb[m][k] * W1[n][k]; one v8i read per B frag
  const v8i* wf1 = (const v8i*)wfrag1;
  v8f acc[4] = {};
  #pragma unroll
  for (int kb = 0; kb < 2; ++kb) {
    #pragma unroll
    for (int nb = 0; nb < 4; ++nb) {
      IV u; u.i = wf1[(kb * 4 + nb) * 32 + lane];
      acc[nb] = __builtin_amdgcn_wmma_f32_16x16x32_f16(
          false, a[kb], false, u.h, (short)0, acc[nb], false, false);
    }
  }

  // --- bias + relu, spill h (C layout: row = v + 8*half, col = nb*16 + mrow)
  _Float16* hb = hbuf[wave];
  #pragma unroll
  for (int nb = 0; nb < 4; ++nb) {
    const int cidx = nb * 16 + mrow;
    const float bb = b1s[cidx];
    #pragma unroll
    for (int v = 0; v < 8; ++v) {
      const int r = v + 8 * half;
      hb[r * 64 + cidx] = (_Float16)fmaxf(acc[nb][v] + bb, 0.0f);
    }
  }
  asm volatile("s_wait_dscnt 0" ::: "memory");   // wave-local LDS RAW

  // --- rebuild A fragments of h: two aligned 16B chunks per fragment
  const int4* hq = (const int4*)hb;
  v16h a2[2];
  #pragma unroll
  for (int kb = 0; kb < 2; ++kb) {
    QV u2;
    const int base = mrow * 8 + kb * 4 + half;   // int4 units (8 halves)
    u2.q[0] = hq[base];       // K = kb*32 + half*8 + 0..7
    u2.q[1] = hq[base + 2];   // K = kb*32 + half*8 + 16..23
    a2[kb] = u2.h;
  }

  // --- GEMM2: logits = h @ W2^T
  const v8i* wf2 = (const v8i*)wfrag2;
  v8f acc2[4] = {};
  #pragma unroll
  for (int kb = 0; kb < 2; ++kb) {
    #pragma unroll
    for (int nb = 0; nb < 4; ++nb) {
      IV u; u.i = wf2[(kb * 4 + nb) * 32 + lane];
      acc2[nb] = __builtin_amdgcn_wmma_f32_16x16x32_f16(
          false, a2[kb], false, u.h, (short)0, acc2[nb], false, false);
    }
  }

  // --- bias + exp + store (padded buffer sized NTILES*16 rows)
  #pragma unroll
  for (int nb = 0; nb < 4; ++nb) {
    const int cidx = nb * 16 + mrow;
    const float bb = b2s[cidx];
    #pragma unroll
    for (int v = 0; v < 8; ++v) {
      const int r = tile * 16 + v + 8 * half;
      elog[(size_t)r * 64 + cidx] = __expf(acc2[nb][v] + bb);
    }
  }
}

// ---------------------------------------------------------------------------
// Kernel 2: zero softmax accumulators and the offset half of d_out
// ---------------------------------------------------------------------------
__global__ __launch_bounds__(256)
void init_kernel(float* __restrict__ num, float* __restrict__ den,
                 float* __restrict__ out_off)
{
  const int i = blockIdx.x * blockDim.x + threadIdx.x;
  if (i < NVIS * 64) { num[i] = 0.0f; den[i] = 0.0f; out_off[i] = 0.0f; }
}

// ---------------------------------------------------------------------------
// Kernel 3: per-edge softmax accumulation (graph1, heads < NV only; softmax
// rows >= NV are discarded by the reference). exp() was precomputed per node,
// so this pass is pure b128 gathers + native f32 atomics.
// 16 lanes per edge, 4 channels per lane (float4).
// ---------------------------------------------------------------------------
__global__ __launch_bounds__(256)
void edge_center_kernel(const int* __restrict__ head, const int* __restrict__ tail,
                        const float* __restrict__ vc, const float* __restrict__ cc,
                        const float* __restrict__ ic, const float* __restrict__ elog,
                        float* __restrict__ num, float* __restrict__ den, int E)
{
  const int gid = blockIdx.x * blockDim.x + threadIdx.x;
  const int e = gid >> 4;
  const int sub = gid & 15;
  if (e >= E) return;
  const int h = head[e];
  if (h >= NVIS) return;
  const int tl = tail[e];
  const float* ep = resolve_row(tl, vc, cc, ic);
  const int c = sub * 4;
  const float4 w4 = *(const float4*)(elog + (size_t)tl * 64 + c);
  const float4 m4 = *(const float4*)(ep + c);
  float* np = num + (size_t)h * 64 + c;
  float* dp = den + (size_t)h * 64 + c;
  atomicAdd(np + 0, w4.x * m4.x);
  atomicAdd(np + 1, w4.y * m4.y);
  atomicAdd(np + 2, w4.z * m4.z);
  atomicAdd(np + 3, w4.w * m4.w);
  atomicAdd(dp + 0, w4.x);
  atomicAdd(dp + 1, w4.y);
  atomicAdd(dp + 2, w4.z);
  atomicAdd(dp + 3, w4.w);
}

// ---------------------------------------------------------------------------
// Kernel 4: vca1 = l2norm(num/den); 16 lanes per visit, width-16 shfl reduce.
// visit_final_emb == vca1 since lam == 1.0 in the reference.
// ---------------------------------------------------------------------------
__global__ __launch_bounds__(256)
void finalize_emb_kernel(const float* __restrict__ num, const float* __restrict__ den,
                         float* __restrict__ out)
{
  const int gid = blockIdx.x * blockDim.x + threadIdx.x;
  const int v = gid >> 4;
  const int sub = gid & 15;
  if (v >= NVIS) return;
  const int c = sub * 4;
  const float4 n4 = *(const float4*)(num + (size_t)v * 64 + c);
  const float4 d4 = *(const float4*)(den + (size_t)v * 64 + c);
  const float x0 = d4.x > 0.0f ? n4.x / d4.x : 0.0f;
  const float x1 = d4.y > 0.0f ? n4.y / d4.y : 0.0f;
  const float x2 = d4.z > 0.0f ? n4.z / d4.z : 0.0f;
  const float x3 = d4.w > 0.0f ? n4.w / d4.w : 0.0f;
  float sq = x0 * x0 + x1 * x1 + x2 * x2 + x3 * x3;
  #pragma unroll
  for (int off = 8; off > 0; off >>= 1)
    sq += __shfl_xor(sq, off, 16);
  const float nrm = sqrtf(fmaxf(sq, 1e-24f));
  const float inv = 1.0f / fmaxf(nrm, 1e-12f);
  float4 o4; o4.x = x0 * inv; o4.y = x1 * inv; o4.z = x2 * inv; o4.w = x3 * inv;
  *(float4*)(out + (size_t)v * 64 + c) = o4;
}

// ---------------------------------------------------------------------------
// Kernel 5: fused offset segment-max. All five reference segment-maxes clamp
// at 0 and values are relu'd (>=0), so IEEE float order == u32 order and a
// single atomicMax(u32) into d_out (pre-zeroed) computes the element-wise max
// over the union of masked edge sets.
// ---------------------------------------------------------------------------
__global__ __launch_bounds__(256)
void edge_offset_kernel(const int* __restrict__ head, const int* __restrict__ tail,
                        const float* __restrict__ vo, const float* __restrict__ co,
                        const float* __restrict__ io, unsigned* __restrict__ out_bits,
                        int E, int need_tail_ge_nv)
{
  const int gid = blockIdx.x * blockDim.x + threadIdx.x;
  const int e = gid >> 4;
  const int sub = gid & 15;
  if (e >= E) return;
  const int h = head[e];
  if (h >= NVIS) return;
  const int tl = tail[e];
  if (need_tail_ge_nv && tl < NVIS) return;   // graph1: both masks need tail >= NV
  const float* op = resolve_row(tl, vo, co, io);
  const int c = sub * 4;
  const float4 o4 = *(const float4*)(op + c);
  unsigned* bp = out_bits + (size_t)h * 64 + c;
  const float r0 = fmaxf(o4.x, 0.0f);
  const float r1 = fmaxf(o4.y, 0.0f);
  const float r2 = fmaxf(o4.z, 0.0f);
  const float r3 = fmaxf(o4.w, 0.0f);
  if (r0 > 0.0f) atomicMax(bp + 0, __float_as_uint(r0));
  if (r1 > 0.0f) atomicMax(bp + 1, __float_as_uint(r1));
  if (r2 > 0.0f) atomicMax(bp + 2, __float_as_uint(r2));
  if (r3 > 0.0f) atomicMax(bp + 3, __float_as_uint(r3));
}

// ---------------------------------------------------------------------------
extern "C" void kernel_launch(void* const* d_in, const int* in_sizes, int n_in,
                              void* d_out, int out_size, void* d_ws, size_t ws_size,
                              hipStream_t stream)
{
  (void)in_sizes; (void)n_in; (void)out_size; (void)ws_size;

  const float* vc = (const float*)d_in[0];   // visit_center  [NV,64]
  const float* vo = (const float*)d_in[1];   // visit_offset  [NV,64]
  const float* cc = (const float*)d_in[2];   // ccs_center    [NC,64]
  const float* co = (const float*)d_in[3];   // ccs_offset    [NC,64]
  const float* ic = (const float*)d_in[4];   // icd_center    [NI,64]
  const float* io = (const float*)d_in[5];   // icd_offset    [NI,64]
  const float* w1 = (const float*)d_in[7];   // att_w1 [64,64]
  const float* b1 = (const float*)d_in[8];   // att_b1 [64]
  const float* w2 = (const float*)d_in[9];   // att_w2 [64,64]
  const float* b2 = (const float*)d_in[10];  // att_b2 [64]
  const int* head1 = (const int*)d_in[15];
  const int* tail1 = (const int*)d_in[16];
  const int* head2 = (const int*)d_in[17];
  const int* tail2 = (const int*)d_in[18];
  // d_in[6], d_in[11..14] (visit_time, tw*/tb*) are dead: lam == 1.0.

  float* out_emb = (float*)d_out;               // visit_final_emb    [NV,64]
  float* out_off = out_emb + (size_t)NVIS * 64; // visit_final_offset [NV,64]

  float* elog = (float*)d_ws;                        // [NTILES*16, 64]
  float* num = elog + (size_t)NTILES * 16 * 64;      // [NV, 64]
  float* den = num + (size_t)NVIS * 64;              // [NV, 64]

  init_kernel<<<(NVIS * 64 + 255) / 256, 256, 0, stream>>>(num, den, out_off);
  node_elog_kernel<<<(NTILES + 7) / 8, 256, 0, stream>>>(vc, cc, ic, w1, b1, w2, b2, elog);
  edge_center_kernel<<<(NE1 * 16) / 256, 256, 0, stream>>>(head1, tail1, vc, cc, ic,
                                                           elog, num, den, NE1);
  finalize_emb_kernel<<<(NVIS * 16 + 255) / 256, 256, 0, stream>>>(num, den, out_emb);
  edge_offset_kernel<<<(NE1 * 16) / 256, 256, 0, stream>>>(head1, tail1, vo, co, io,
                                                           (unsigned*)out_off, NE1, 1);
  edge_offset_kernel<<<(NE2 * 16) / 256, 256, 0, stream>>>(head2, tail2, vo, co, io,
                                                           (unsigned*)out_off, NE2, 0);
}
